// GRU_Policy_32719060861680
// MI455X (gfx1250) — matrix operand order repaired and checked by reading, not compiled, not run on hardware
//
#include <hip/hip_runtime.h>

#define S_LEN 2048
#define BATCH 256
#define DIN   64
#define HID   256
#define G3    768
#define OUTD  32
#define BT    16            // batch rows per workgroup (WMMA M)
#define NWG   (BATCH/BT)    // 16 workgroups
#define NTHR  512           // 16 waves
#define HPAD  264           // LDS row stride (halves) -> conflict-free b128 A loads
#define XPAD  72

typedef __attribute__((ext_vector_type(16))) _Float16 v16h;
typedef __attribute__((ext_vector_type(8)))  _Float16 v8h;
typedef __attribute__((ext_vector_type(2)))  _Float16 v2h;
typedef __attribute__((ext_vector_type(8)))  float    v8f;

__device__ __forceinline__ v16h cat8(v8h lo, v8h hi) {
  return __builtin_shufflevector(lo, hi, 0,1,2,3,4,5,6,7,8,9,10,11,12,13,14,15);
}

// B-matrix 32x16 f16 fragment, address = Wbase + voff(+k0 folded into imm offset).
// voff encodes: (col_base + lane%16)*K + (lane<16 ? 0 : 8)
__device__ __forceinline__ v16h load_b_frag(const _Float16* __restrict__ Wbase,
                                            int voff, int k0) {
  const v8h* p = (const v8h*)(Wbase + voff + k0);
  return cat8(p[0], p[2]);
}

// A-matrix 16x32 f16 fragment from LDS (row-major, padded stride).
// lane m<16: K k0..k0+7 & k0+16..k0+23 ; lane m+16: k0+8..15 & k0+24..31
__device__ __forceinline__ v16h load_a_frag(const _Float16* sh, int stride,
                                            int k0, int lane) {
  const int m   = lane & 15;
  const int off = (lane < 16) ? 0 : 8;
  const v8h* p = (const v8h*)(sh + m * stride + k0 + off);
  const v8h* q = (const v8h*)(sh + m * stride + k0 + off + 16);
  return cat8(*p, *q);
}

__device__ __forceinline__ v8f wmma_f16(v16h a, v16h b, v8f c) {
  return __builtin_amdgcn_wmma_f32_16x16x32_f16(false, a, false, b, (short)0, c,
                                                false, false);
}

__device__ __forceinline__ float sigm(float x) { return 1.f / (1.f + __expf(-x)); }
__device__ __forceinline__ float tanh_fast(float x) {
  x = fminf(15.f, fmaxf(-15.f, x));
  const float e = __expf(-2.f * x);
  return (1.f - e) / (1.f + e);
}

__global__ void cvt_f32_to_f16(const float* __restrict__ src,
                               _Float16* __restrict__ dst, int n) {
  int i = blockIdx.x * blockDim.x + threadIdx.x;
  if (i < n) dst[i] = (_Float16)src[i];
}

__global__ void __launch_bounds__(NTHR, 1)
gru_fused(const float* __restrict__ x,
          const _Float16* __restrict__ Wih0_, const _Float16* __restrict__ Whh0_,
          const float* __restrict__ bih0,  const float* __restrict__ bhh0,
          const _Float16* __restrict__ Wih1_, const _Float16* __restrict__ Whh1_,
          const float* __restrict__ bih1,  const float* __restrict__ bhh1,
          const _Float16* __restrict__ Wout, const float* __restrict__ bout,
          float* __restrict__ out) {
  __shared__ _Float16 shX [BT * XPAD];
  __shared__ _Float16 shH0[BT * HPAD];
  __shared__ _Float16 shH1[BT * HPAD];

  const int tid   = threadIdx.x;
  const int lane  = tid & 31;
  const int wave  = tid >> 5;              // 0..15, owns h-cols [wave*16, wave*16+16)
  const int bbase = blockIdx.x * BT;

  for (int i = tid; i < BT * HPAD; i += NTHR) {
    shH0[i] = (_Float16)0.f;
    shH1[i] = (_Float16)0.f;
  }

  // per-lane combined bias scalars (r/z summed, n split: n = tanh((xn+bin)+r*(hn+bhn)))
  const int ccol = wave * 16 + (lane & 15);
  const float br0  = bih0[0 * HID + ccol] + bhh0[0 * HID + ccol];
  const float bz0  = bih0[1 * HID + ccol] + bhh0[1 * HID + ccol];
  const float bxn0 = bih0[2 * HID + ccol];
  const float bhn0 = bhh0[2 * HID + ccol];
  const float br1  = bih1[0 * HID + ccol] + bhh1[0 * HID + ccol];
  const float bz1  = bih1[1 * HID + ccol] + bhh1[1 * HID + ccol];
  const float bxn1 = bih1[2 * HID + ccol];
  const float bhn1 = bhh1[2 * HID + ccol];

  v8f h0r = {}, h1r = {};                  // h fragments (C layout), this wave's 16 cols
  const int hm   = (lane < 16) ? 0 : 8;    // C layout: vgpr i -> row i+hm, col lane&15
  const int hcol = wave * 16 + (lane & 15);

  // loop-invariant per-lane B-fragment offsets (elements); k-step folds into imm
  const int boff  = (lane < 16) ? 0 : 8;
  const int cR    = wave * 16 + (lane & 15);
  const int voxR  = (cR)           * DIN + boff;   // K=64 matrix (Wih0)
  const int voxZ  = (cR + HID)     * DIN + boff;
  const int voxN  = (cR + 2 * HID) * DIN + boff;
  const int vohR  = (cR)           * HID + boff;   // K=256 matrices (Whh0/Wih1/Whh1)
  const int vohZ  = (cR + HID)     * HID + boff;
  const int vohN  = (cR + 2 * HID) * HID + boff;

  __syncthreads();

  size_t tweak = 0;                        // opaque 0, redefined each iteration:
                                           // keeps weight loads in-loop (no LICM/spill)
                                           // while preserving global provenance
  for (int s = 0; s < S_LEN; ++s) {
    asm volatile("" : "+s"(tweak));
    const _Float16* Wih0 = Wih0_ + tweak;
    const _Float16* Whh0 = Whh0_ + tweak;
    const _Float16* Wih1 = Wih1_ + tweak;
    const _Float16* Whh1 = Whh1_ + tweak;

    // ---- stage x[s] batch-tile into LDS as f16 (512 thr * 2 floats = 16x64) ----
    {
      const float* xs = x + ((size_t)s * BATCH + bbase) * DIN;
      const int row = tid >> 5;              // 0..15
      const int col = (tid & 31) * 2;        // 0..62 (even -> 4B-aligned ds_store_b32)
      const float2 v = *(const float2*)(xs + row * DIN + col);
      v2h pk; pk[0] = (_Float16)v.x; pk[1] = (_Float16)v.y;
      *(v2h*)&shX[row * XPAD + col] = pk;
      if (s + 1 < S_LEN)
        __builtin_prefetch(xs + (size_t)BATCH * DIN + row * DIN + col, 0, 1);
    }
    __syncthreads();

    // ---- layer 0: cr/cz accumulate BOTH x- and h-GEMMs; cxn/chn kept split ----
    v8f cr = {}, cz = {}, cxn = {}, chn = {};
#pragma unroll
    for (int ks = 0; ks < DIN / 32; ++ks) {
      const v16h a = load_a_frag(shX, XPAD, ks * 32, lane);
      cr  = wmma_f16(a, load_b_frag(Wih0, voxR, ks * 32), cr);
      cz  = wmma_f16(a, load_b_frag(Wih0, voxZ, ks * 32), cz);
      cxn = wmma_f16(a, load_b_frag(Wih0, voxN, ks * 32), cxn);
    }
#pragma unroll
    for (int ks = 0; ks < HID / 32; ++ks) {
      const v16h a = load_a_frag(shH0, HPAD, ks * 32, lane);
      cr  = wmma_f16(a, load_b_frag(Whh0, vohR, ks * 32), cr);
      cz  = wmma_f16(a, load_b_frag(Whh0, vohZ, ks * 32), cz);
      chn = wmma_f16(a, load_b_frag(Whh0, vohN, ks * 32), chn);
    }
    __syncthreads();                         // all reads of shH0 done

    // ---- gates layer 0 (register-local), update h0 regs + LDS mirror ----
#pragma unroll
    for (int i = 0; i < 8; ++i) {
      const float r  = sigm(cr[i] + br0);
      const float z  = sigm(cz[i] + bz0);
      const float n  = tanh_fast(cxn[i] + bxn0 + r * (chn[i] + bhn0));
      const float hv = (1.f - z) * n + z * h0r[i];
      h0r[i] = hv;
      shH0[(i + hm) * HPAD + hcol] = (_Float16)hv;
    }
    __syncthreads();                         // shH0 now holds h0[s]

    // ---- layer 1 (two separate K-loops to keep fragment pressure low) ----
    cr = v8f{}; cz = v8f{}; cxn = v8f{}; chn = v8f{};
#pragma unroll
    for (int ks = 0; ks < HID / 32; ++ks) {
      const v16h a = load_a_frag(shH0, HPAD, ks * 32, lane);
      cr  = wmma_f16(a, load_b_frag(Wih1, vohR, ks * 32), cr);
      cz  = wmma_f16(a, load_b_frag(Wih1, vohZ, ks * 32), cz);
      cxn = wmma_f16(a, load_b_frag(Wih1, vohN, ks * 32), cxn);
    }
#pragma unroll
    for (int ks = 0; ks < HID / 32; ++ks) {
      const v16h a = load_a_frag(shH1, HPAD, ks * 32, lane);
      cr  = wmma_f16(a, load_b_frag(Whh1, vohR, ks * 32), cr);
      cz  = wmma_f16(a, load_b_frag(Whh1, vohZ, ks * 32), cz);
      chn = wmma_f16(a, load_b_frag(Whh1, vohN, ks * 32), chn);
    }
    __syncthreads();                         // all reads of shH1 done

#pragma unroll
    for (int i = 0; i < 8; ++i) {
      const float r  = sigm(cr[i] + br1);
      const float z  = sigm(cz[i] + bz1);
      const float n  = tanh_fast(cxn[i] + bxn1 + r * (chn[i] + bhn1));
      const float hv = (1.f - z) * n + z * h1r[i];
      h1r[i] = hv;
      shH1[(i + hm) * HPAD + hcol] = (_Float16)hv;
    }
    // next iteration's first barrier publishes shH1 before it is read again
  }
  __syncthreads();

  // ---- output head: out[layer] = tanh(hT @ Wout^T + bout), (16x256)@(256x32) ----
  if (wave < 4) {
    const int layer = wave >> 1;
    const int tile  = wave & 1;
    const _Float16* shH = layer ? shH1 : shH0;
    const int ocol = tile * 16 + (lane & 15);
    const int voO  = ocol * HID + boff;
    v8f c = {};
#pragma unroll
    for (int ks = 0; ks < HID / 32; ++ks) {
      const v16h a = load_a_frag(shH, HPAD, ks * 32, lane);
      c = wmma_f16(a, load_b_frag(Wout, voO, ks * 32), c);
    }
    const float bo = bout[ocol];
#pragma unroll
    for (int i = 0; i < 8; ++i) {
      const int row = bbase + i + hm;
      out[(size_t)layer * BATCH * OUTD + row * OUTD + ocol] = tanh_fast(c[i] + bo);
    }
  }
}

extern "C" void kernel_launch(void* const* d_in, const int* in_sizes, int n_in,
                              void* d_out, int out_size, void* d_ws, size_t ws_size,
                              hipStream_t stream) {
  const float* x    = (const float*)d_in[0];
  const float* Wih0 = (const float*)d_in[1];
  const float* Whh0 = (const float*)d_in[2];
  const float* bih0 = (const float*)d_in[3];
  const float* bhh0 = (const float*)d_in[4];
  const float* Wih1 = (const float*)d_in[5];
  const float* Whh1 = (const float*)d_in[6];
  const float* bih1 = (const float*)d_in[7];
  const float* bhh1 = (const float*)d_in[8];
  const float* Wout = (const float*)d_in[9];
  const float* bout = (const float*)d_in[10];

  // workspace layout: f16 copies of the weight matrices (~1.3 MB total)
  _Float16* hWih0 = (_Float16*)d_ws;
  _Float16* hWhh0 = hWih0 + G3 * DIN;
  _Float16* hWih1 = hWhh0 + G3 * HID;
  _Float16* hWhh1 = hWih1 + G3 * HID;
  _Float16* hWout = hWhh1 + G3 * HID;

  const int nIh0 = G3 * DIN, nH = G3 * HID, nO = OUTD * HID;
  cvt_f32_to_f16<<<(nIh0 + 255) / 256, 256, 0, stream>>>(Wih0, hWih0, nIh0);
  cvt_f32_to_f16<<<(nH   + 255) / 256, 256, 0, stream>>>(Whh0, hWhh0, nH);
  cvt_f32_to_f16<<<(nH   + 255) / 256, 256, 0, stream>>>(Wih1, hWih1, nH);
  cvt_f32_to_f16<<<(nH   + 255) / 256, 256, 0, stream>>>(Whh1, hWhh1, nH);
  cvt_f32_to_f16<<<(nO   + 255) / 256, 256, 0, stream>>>(Wout, hWout, nO);

  gru_fused<<<NWG, NTHR, 0, stream>>>(x, hWih0, hWhh0, bih0, bhh0,
                                      hWih1, hWhh1, bih1, bhh1,
                                      hWout, bout, (float*)d_out);
}
